// AttentiveAtlasEncoder_27779848471402
// MI455X (gfx1250) — matrix operand order repaired and checked by reading, not compiled, not run on hardware
//
#include <hip/hip_runtime.h>
#include <hip/hip_bf16.h>

typedef __attribute__((ext_vector_type(16))) _Float16 v16h;
typedef __attribute__((ext_vector_type(8)))  float    v8f;

constexpr int Bn = 65536;
constexpr int DIN = 64;
constexpr int Hd = 512;
constexpr int Ld = 64;
constexpr int NCn = 8;
constexpr int CODESn = 256;
constexpr int LHd = 32;

// flat output offsets (return-order concat, all stored as f32)
constexpr size_t OFF_KCHART = 0;
constexpr size_t OFF_KCODE  = (size_t)Bn;
constexpr size_t OFF_ZN     = (size_t)2 * Bn;
constexpr size_t OFF_ZTEX   = OFF_ZN + (size_t)Bn * Ld;
constexpr size_t OFF_ROUTER = OFF_ZTEX + (size_t)Bn * Ld;
constexpr size_t OFF_ZGEO   = OFF_ROUTER + (size_t)Bn * NCn;
constexpr size_t OFF_VQ     = OFF_ZGEO + (size_t)Bn * Ld;
constexpr size_t OFF_ISTACK = OFF_VQ + 1;
constexpr size_t OFF_ZNALL  = OFF_ISTACK + (size_t)Bn * NCn;

__device__ __forceinline__ float gelu_f(float x) {
  return 0.5f * x * (1.0f + erff(x * 0.70710678118654752440f));
}

// ---- WMMA fragment packing per CDNA5 ISA VGPR layout tables ----
// A fragment (16x32 f16): lane = m + 16*((k>>3)&1); elem = 2*(((k>>4)<<2)|((k>>1)&3)) + (k&1)
__device__ __forceinline__ int a_pack_idx(int m, int k) {
  int hf  = (k >> 3) & 1;
  int i   = ((k >> 4) << 2) | ((k >> 1) & 3);
  return (m + 16 * hf) * 16 + 2 * i + (k & 1);
}

// Packed fragment load: one v16h (32B) per lane, contiguous -> 2x b128 loads
__device__ __forceinline__ v16h load_frag(const _Float16* base, int lane) {
  return *(const v16h*)(base + lane * 16);
}

#define WMMA_F16(a, b, c) \
  __builtin_amdgcn_wmma_f32_16x16x32_f16(false, (a), false, (b), (short)0, (c), false, false)

// ---------------- K0a: pack a KxN f32 weight into WMMA-B fragment layout ----------------
// element(k,n) = src[k*sk + n*sn]; dst fragment frag = nt*(K/32)+kt, 512 halves each.
// B layout: lane -> N = lane&15; elem e<8 holds K=8*(lane>>4)+e, e>=8 holds K=16+8*(lane>>4)+(e-8).
__global__ void k_pack_b(const float* __restrict__ src, _Float16* __restrict__ dst,
                         int K, int N, int sk, int sn) {
  int id = blockIdx.x * 256 + threadIdx.x;
  if (id >= K * N) return;
  int e = id & 15, lane = (id >> 4) & 31, frag = id >> 9;
  int ktiles = K >> 5;
  int kt = frag % ktiles, nt = frag / ktiles;
  int i = e >> 1, par = e & 1;
  int k = kt * 32 + ((e >> 3) << 4) + ((lane >> 4) << 3) + (e & 7);
  int n = nt * 16 + (lane & 15);
  dst[id] = (_Float16)src[(size_t)k * sk + (size_t)n * sn];
}

__global__ void k_cbnorm(const float* __restrict__ cb, float* __restrict__ cbn) {
  int i = blockIdx.x * 256 + threadIdx.x;
  if (i < NCn * CODESn) {
    const float* p = cb + (size_t)i * Ld;
    float s = 0.f;
    for (int l = 0; l < Ld; ++l) s += p[l] * p[l];
    cbn[i] = s;
  }
}

// ---------------- K1: x -> h(LN,gelu) -> feat (stored packed-A in global) ----------------
__global__ __launch_bounds__(256) void k_stage1(
    const float* __restrict__ x, const float* __restrict__ b1,
    const float* __restrict__ ln_g, const float* __restrict__ ln_b,
    const float* __restrict__ b2,
    const _Float16* __restrict__ W1p, const _Float16* __restrict__ W2p,
    _Float16* __restrict__ featp) {
  __shared__ __align__(32) _Float16 xs_p[2 * 512];       // x tile, packed A (K=64)
  __shared__ __align__(16) char     hbuf[16 * Hd * 4];   // f32 h, then packed f16 feat
  __shared__ __align__(32) _Float16 ha_p[16 * 512];      // LN+gelu h, packed A (K=512)
  __shared__ float red1[256], red2[256], mu_s[16], rs_s[16];
  float*    hs   = (float*)hbuf;
  _Float16* fp_s = (_Float16*)hbuf;

  const int tid = threadIdx.x, lane = tid & 31, w = tid >> 5;
  const int row0 = blockIdx.x * 16;

  for (int i = tid; i < 16 * DIN; i += 256) {
    int m = i >> 6, col = i & 63;
    xs_p[(col >> 5) * 512 + a_pack_idx(m, col & 31)] = (_Float16)x[(size_t)row0 * DIN + i];
  }
  __syncthreads();

  // GEMM1: h = x @ W1 + b1   (16 x 512, K=64)
  {
    v8f acc[4] = {};
#pragma unroll
    for (int kk = 0; kk < 2; ++kk) {
      v16h a = load_frag(xs_p + kk * 512, lane);
#pragma unroll
      for (int j = 0; j < 4; ++j) {
        v16h b = load_frag(W1p + ((size_t)(w * 4 + j) * 2 + kk) * 512, lane);
        acc[j] = WMMA_F16(a, b, acc[j]);
      }
    }
    int n = lane & 15, hf = lane >> 4;
#pragma unroll
    for (int j = 0; j < 4; ++j) {
      int n0 = (w * 4 + j) * 16;
#pragma unroll
      for (int r = 0; r < 8; ++r)
        hs[(r + 8 * hf) * Hd + n0 + n] = acc[j][r] + b1[n0 + n];
    }
  }
  __syncthreads();

  // LayerNorm statistics per row
  {
    int r = tid >> 4, j = tid & 15;
    float s = 0.f, s2 = 0.f;
    for (int i = 0; i < 32; ++i) {
      float hv = hs[r * Hd + j * 32 + i];
      s += hv; s2 += hv * hv;
    }
    red1[tid] = s; red2[tid] = s2;
  }
  __syncthreads();
  if ((tid & 15) == 0) {
    int r = tid >> 4;
    float a = 0.f, b = 0.f;
    for (int t = 0; t < 16; ++t) { a += red1[r * 16 + t]; b += red2[r * 16 + t]; }
    float mu = a * (1.0f / Hd);
    float var = b * (1.0f / Hd) - mu * mu;
    mu_s[r] = mu;
    rs_s[r] = rsqrtf(var + 1e-5f);
  }
  __syncthreads();
  {
    int r = tid >> 4, j = tid & 15;
    float mu = mu_s[r], rs = rs_s[r];
    for (int i = 0; i < 32; ++i) {
      int col = j * 32 + i;
      float val = (hs[r * Hd + col] - mu) * rs * ln_g[col] + ln_b[col];
      ha_p[(col >> 5) * 512 + a_pack_idx(r, col & 31)] = (_Float16)gelu_f(val);
    }
  }
  __syncthreads();  // after this, hs is dead; hbuf reused as fp_s

  // GEMM2: feat = gelu(h @ W2 + b2)   (16 x 512, K=512), store packed into LDS
  {
    v8f acc[4] = {};
#pragma unroll
    for (int kk = 0; kk < 16; ++kk) {
      v16h a = load_frag(ha_p + kk * 512, lane);
#pragma unroll
      for (int j = 0; j < 4; ++j) {
        v16h b = load_frag(W2p + ((size_t)(w * 4 + j) * 16 + kk) * 512, lane);
        acc[j] = WMMA_F16(a, b, acc[j]);
      }
    }
    int n = lane & 15, hf = lane >> 4;
#pragma unroll
    for (int j = 0; j < 4; ++j) {
      int n0 = (w * 4 + j) * 16;
#pragma unroll
      for (int r = 0; r < 8; ++r) {
        int col = n0 + n, m16 = r + 8 * hf;
        fp_s[(col >> 5) * 512 + a_pack_idx(m16, col & 31)] =
            (_Float16)gelu_f(acc[j][r] + b2[col]);
      }
    }
  }
  __syncthreads();

  // bulk b128 copy of packed feat tile to global
  {
    float4* dst = (float4*)(featp + (size_t)blockIdx.x * 8192);
    const float4* src = (const float4*)fp_s;
    for (int i = tid; i < 1024; i += 256) dst[i] = src[i];
  }
}

// ---------------- K2: router (k, scores, softmax) and v ----------------
__global__ __launch_bounds__(256) void k_router_v(
    const _Float16* __restrict__ featp, const _Float16* __restrict__ Wkp,
    const float* __restrict__ bk, const float* __restrict__ chart_q,
    const _Float16* __restrict__ Wvp, const float* __restrict__ bv,
    float* __restrict__ out, float* __restrict__ v_f) {
  __shared__ __align__(32) _Float16 fa_p[16 * Hd];
  __shared__ float ks[16 * Hd];
  __shared__ float sc[16 * NCn];
  const int tid = threadIdx.x, lane = tid & 31, w = tid >> 5;
  const int row0 = blockIdx.x * 16;

  {
    const float4* src = (const float4*)(featp + (size_t)blockIdx.x * 8192);
    float4* dst = (float4*)fa_p;
    for (int i = tid; i < 1024; i += 256) dst[i] = src[i];
  }
  __syncthreads();

  // k = feat @ Wk + bk
  {
    v8f acc[4] = {};
#pragma unroll
    for (int kk = 0; kk < 16; ++kk) {
      v16h a = load_frag(fa_p + kk * 512, lane);
#pragma unroll
      for (int j = 0; j < 4; ++j) {
        v16h b = load_frag(Wkp + ((size_t)(w * 4 + j) * 16 + kk) * 512, lane);
        acc[j] = WMMA_F16(a, b, acc[j]);
      }
    }
    int n = lane & 15, hf = lane >> 4;
#pragma unroll
    for (int j = 0; j < 4; ++j) {
      int n0 = (w * 4 + j) * 16;
#pragma unroll
      for (int r = 0; r < 8; ++r)
        ks[(r + 8 * hf) * Hd + n0 + n] = acc[j][r] + bk[n0 + n];
    }
  }
  __syncthreads();

  // scores = k @ chart_q^T / sqrt(H)
  if (tid < 128) {
    int r = tid >> 3, ch = tid & 7;
    const float* cq = chart_q + (size_t)ch * Hd;
    float s = 0.f;
    for (int i = 0; i < Hd; ++i) s += ks[r * Hd + i] * cq[i];
    sc[r * NCn + ch] = s * 0.044194173824159216f;  // 1/sqrt(512)
  }
  __syncthreads();

  // softmax + K_chart
  if (tid < 16) {
    float m = -3.4e38f; int am = 0;
    for (int c = 0; c < NCn; ++c) {
      float vv = sc[tid * NCn + c];
      if (vv > m) { m = vv; am = c; }
    }
    float e[NCn]; float se = 0.f;
    for (int c = 0; c < NCn; ++c) { e[c] = __expf(sc[tid * NCn + c] - m); se += e[c]; }
    float inv = 1.f / se;
    size_t row = (size_t)row0 + tid;
    for (int c = 0; c < NCn; ++c) out[OFF_ROUTER + row * NCn + c] = e[c] * inv;
    out[OFF_KCHART + row] = (float)am;
  }

  // v = feat @ Wv + bv   (waves 0..3, one 16-col tile each; wave-uniform branch)
  if (w < 4) {
    v8f acc = {};
#pragma unroll
    for (int kk = 0; kk < 16; ++kk) {
      v16h a = load_frag(fa_p + kk * 512, lane);
      v16h b = load_frag(Wvp + ((size_t)w * 16 + kk) * 512, lane);
      acc = WMMA_F16(a, b, acc);
    }
    int n = lane & 15, hf = lane >> 4;
#pragma unroll
    for (int r = 0; r < 8; ++r) {
      int col = w * 16 + n;
      size_t row = (size_t)row0 + r + 8 * hf;
      v_f[row * Ld + col] = acc[r] + bv[col];
    }
  }
}

// ---------------- K3: VQ search + smoother MLP + outputs ----------------
__global__ __launch_bounds__(256) void k_vq(
    const float* __restrict__ v_f, const _Float16* __restrict__ cbp,
    const float* __restrict__ cbn, const float* __restrict__ cb_f32,
    const _Float16* __restrict__ Ws1p, const float* __restrict__ bs1,
    const _Float16* __restrict__ Ws2p, const float* __restrict__ bs2,
    float* __restrict__ out, float* __restrict__ vqPart) {
  __shared__ float    vs[16 * Ld];
  __shared__ __align__(32) _Float16 va_p[2 * 512];      // v packed A (K=64)
  __shared__ float    vnorm[16];
  __shared__ float    rtile[16 * NCn];
  __shared__ int      idxs[16 * NCn];
  __shared__ float    redv[8 * 16];
  __shared__ int      redi[8 * 16];
  __shared__ __align__(32) _Float16 dlt_p[16 * 512];    // 8 row-tiles x 2 k-tiles, packed A
  __shared__ __align__(32) _Float16 t1_p[8 * 512];      // 8 row-tiles, packed A (K=32)
  __shared__ float    zqb[16 * Ld];
  __shared__ float    sred[256];

  const int tid = threadIdx.x, lane = tid & 31, w = tid >> 5;
  const int row0 = blockIdx.x * 16;

  for (int i = tid; i < 16 * Ld; i += 256) {
    float vv = v_f[(size_t)row0 * Ld + i];
    vs[i] = vv;
    int m = i >> 6, col = i & 63;
    va_p[(col >> 5) * 512 + a_pack_idx(m, col & 31)] = (_Float16)vv;
  }
  for (int i = tid; i < 16 * NCn; i += 256)
    rtile[i] = out[OFF_ROUTER + (size_t)row0 * NCn + i];
  __syncthreads();
  if (tid < 16) {
    float s = 0.f;
    for (int l = 0; l < Ld; ++l) { float t = vs[tid * Ld + l]; s += t * t; }
    vnorm[tid] = s;
  }
  __syncthreads();

  // codebook argmin per chart via WMMA dot products (v @ cb^T); cbp: K=64, N=2048 packed
  for (int nc = 0; nc < NCn; ++nc) {
    float best[8]; int bidx[8];
#pragma unroll
    for (int r = 0; r < 8; ++r) { best[r] = 3.4e38f; bidx[r] = 0; }
#pragma unroll
    for (int j = 0; j < 2; ++j) {
      int nt = nc * 16 + w * 2 + j;
      v8f acc = {};
#pragma unroll
      for (int kk = 0; kk < 2; ++kk) {
        v16h a = load_frag(va_p + kk * 512, lane);
        v16h b = load_frag(cbp + ((size_t)nt * 2 + kk) * 512, lane);
        acc = WMMA_F16(a, b, acc);
      }
      int n = lane & 15, hf = lane >> 4;
      int n0 = (w * 2 + j) * 16;
#pragma unroll
      for (int r = 0; r < 8; ++r) {
        int m = r + 8 * hf, c = n0 + n;
        float d2 = vnorm[m] + cbn[nc * CODESn + c] - 2.f * acc[r];
        if (d2 < best[r]) { best[r] = d2; bidx[r] = c; }
      }
    }
#pragma unroll
    for (int off = 8; off; off >>= 1) {
#pragma unroll
      for (int r = 0; r < 8; ++r) {
        float ov = __shfl_xor(best[r], off, 32);
        int   oi = __shfl_xor(bidx[r], off, 32);
        if (ov < best[r]) { best[r] = ov; bidx[r] = oi; }
      }
    }
    if ((lane & 15) == 0) {
      int hf = lane >> 4;
      for (int r = 0; r < 8; ++r) {
        redv[w * 16 + r + 8 * hf] = best[r];
        redi[w * 16 + r + 8 * hf] = bidx[r];
      }
    }
    __syncthreads();
    if (tid < 16) {
      float bv2 = 3.4e38f; int bi = 0;
      for (int ww = 0; ww < 8; ++ww) {
        float vv2 = redv[ww * 16 + tid];
        if (vv2 < bv2) { bv2 = vv2; bi = redi[ww * 16 + tid]; }
      }
      idxs[tid * NCn + nc] = bi;
    }
    __syncthreads();
  }

  // K_code + indices_stack
  if (tid < 16) {
    size_t row = (size_t)row0 + tid;
    float m = -3.4e38f; int am = 0;
    for (int c = 0; c < NCn; ++c) {
      float rv = rtile[tid * NCn + c];
      if (rv > m) { m = rv; am = c; }
    }
    out[OFF_KCODE + row] = (float)idxs[tid * NCn + am];
    for (int c = 0; c < NCn; ++c)
      out[OFF_ISTACK + row * NCn + c] = (float)idxs[tid * NCn + c];
  }

  // delta (packed A), z_q_blended, vq-loss partial (deterministic)
  float lsum = 0.f;
  for (int e = tid; e < 16 * Ld; e += 256) {
    int lr = e >> 6, col = e & 63;
    float vv = vs[e];
    float zsum = 0.f;
#pragma unroll
    for (int nc = 0; nc < NCn; ++nc) {
      int ci = idxs[lr * NCn + nc];
      float zq = cb_f32[((size_t)nc * CODESn + ci) * Ld + col];
      float d = vv - zq;
      int dr = lr * NCn + nc;  // delta-row in [0,128)
      dlt_p[((dr >> 4) * 2 + (col >> 5)) * 512 + a_pack_idx(dr & 15, col & 31)] = (_Float16)d;
      float ww = rtile[lr * NCn + nc];
      zsum += ww * zq;
      lsum += ww * d * d;  // forward value: commitment == codebook term
    }
    zqb[e] = zsum;
  }
  sred[tid] = lsum;
  __syncthreads();
  for (int s = 128; s; s >>= 1) {
    if (tid < s) sred[tid] += sred[tid + s];
    __syncthreads();
  }
  if (tid == 0) vqPart[blockIdx.x] = sred[0];
  __syncthreads();

  // smoother GEMM1: t1 = gelu(delta @ Ws1 + bs1)  (128 x 32, K=64); Ws1p: K=64,N=32
  {
    int rt = w;
#pragma unroll
    for (int ct = 0; ct < 2; ++ct) {
      v8f acc = {};
#pragma unroll
      for (int kk = 0; kk < 2; ++kk) {
        v16h a = load_frag(dlt_p + (rt * 2 + kk) * 512, lane);
        v16h b = load_frag(Ws1p + ((size_t)ct * 2 + kk) * 512, lane);
        acc = WMMA_F16(a, b, acc);
      }
      int n = lane & 15, hf = lane >> 4;
#pragma unroll
      for (int r = 0; r < 8; ++r) {
        int col = ct * 16 + n, m16 = r + 8 * hf;
        t1_p[rt * 512 + a_pack_idx(m16, col)] = (_Float16)gelu_f(acc[r] + bs1[col]);
      }
    }
  }
  __syncthreads();

  // smoother GEMM2: z_n_all = t1 @ Ws2 + bs2  (128 x 64, K=32); Ws2p: K=32,N=64
  {
#pragma unroll
    for (int j = 0; j < 4; ++j) {
      int idx = w * 4 + j, rt = idx >> 2, ct = idx & 3;
      v8f acc = {};
      v16h a = load_frag(t1_p + rt * 512, lane);
      v16h b = load_frag(Ws2p + (size_t)ct * 512, lane);
      acc = WMMA_F16(a, b, acc);
      int n = lane & 15, hf = lane >> 4;
#pragma unroll
      for (int r = 0; r < 8; ++r) {
        int m16 = r + 8 * hf;
        int dr = rt * 16 + m16, lr = dr >> 3, nc = dr & 7, col = ct * 16 + n;
        float val = acc[r] + bs2[col];
        out[OFF_ZNALL + (((size_t)row0 + lr) * NCn + nc) * Ld + col] = val;
      }
    }
  }
  __syncthreads();

  // final outputs: z_n, z_tex, z_geo (reads back this block's z_n_all)
  for (int e = tid; e < 16 * Ld; e += 256) {
    int lr = e >> 6, col = e & 63;
    size_t row = (size_t)row0 + lr;
    float znv = 0.f;
#pragma unroll
    for (int nc = 0; nc < NCn; ++nc)
      znv += rtile[lr * NCn + nc] * out[OFF_ZNALL + (row * NCn + nc) * Ld + col];
    float zq = zqb[e], vv = vs[e];
    out[OFF_ZN + row * Ld + col]   = znv;
    out[OFF_ZTEX + row * Ld + col] = vv - zq - znv;
    out[OFF_ZGEO + row * Ld + col] = zq + znv;
  }
}

// ---------------- K4: finish vq_loss ----------------
__global__ void k_finish(const float* __restrict__ vqPart, float* __restrict__ out) {
  float s = 0.f;
  for (int i = 0; i < Bn / 16; ++i) s += vqPart[i];
  out[OFF_VQ] = 1.25f * s / ((float)Bn * (float)Ld);
}

extern "C" void kernel_launch(void* const* d_in, const int* in_sizes, int n_in,
                              void* d_out, int out_size, void* d_ws, size_t ws_size,
                              hipStream_t stream) {
  const float* x       = (const float*)d_in[0];
  const float* W1      = (const float*)d_in[1];
  const float* b1      = (const float*)d_in[2];
  const float* ln_g    = (const float*)d_in[3];
  const float* ln_b    = (const float*)d_in[4];
  const float* W2      = (const float*)d_in[5];
  const float* b2      = (const float*)d_in[6];
  const float* Wk      = (const float*)d_in[7];
  const float* bk      = (const float*)d_in[8];
  const float* chart_q = (const float*)d_in[9];
  const float* Wv      = (const float*)d_in[10];
  const float* bv      = (const float*)d_in[11];
  const float* cb      = (const float*)d_in[12];
  const float* Ws1     = (const float*)d_in[13];
  const float* bs1     = (const float*)d_in[14];
  const float* Ws2     = (const float*)d_in[15];
  const float* bs2     = (const float*)d_in[16];
  float* out = (float*)d_out;

  char* p = (char*)d_ws;
  auto carve = [&](size_t bytes) {
    char* r = p;
    p += (bytes + 255) & ~(size_t)255;
    return r;
  };
  _Float16* W1p   = (_Float16*)carve((size_t)DIN * Hd * 2);
  _Float16* W2p   = (_Float16*)carve((size_t)Hd * Hd * 2);
  _Float16* Wkp   = (_Float16*)carve((size_t)Hd * Hd * 2);
  _Float16* Wvp   = (_Float16*)carve((size_t)Hd * Ld * 2);
  _Float16* Ws1p  = (_Float16*)carve((size_t)Ld * LHd * 2);
  _Float16* Ws2p  = (_Float16*)carve((size_t)LHd * Ld * 2);
  _Float16* cbp   = (_Float16*)carve((size_t)NCn * CODESn * Ld * 2);
  float*    cbn   = (float*)carve((size_t)NCn * CODESn * 4);
  float*    vqPart= (float*)carve((size_t)(Bn / 16) * 4);
  _Float16* featp = (_Float16*)carve((size_t)Bn * Hd * 2);
  float*    v_f   = (float*)carve((size_t)Bn * Ld * 4);

  auto pack = [&](const float* s, _Float16* d, int K, int N, int sk, int sn) {
    k_pack_b<<<(K * N + 255) / 256, 256, 0, stream>>>(s, d, K, N, sk, sn);
  };
  pack(W1,  W1p,  DIN, Hd, Hd, 1);
  pack(W2,  W2p,  Hd,  Hd, Hd, 1);
  pack(Wk,  Wkp,  Hd,  Hd, Hd, 1);
  pack(Wv,  Wvp,  Hd,  Ld, Ld, 1);
  pack(Ws1, Ws1p, Ld,  LHd, LHd, 1);
  pack(Ws2, Ws2p, LHd, Ld, Ld, 1);
  // codebooks as B^T: element(k, n=global code) = cb[n*64 + k]
  pack(cb,  cbp,  Ld, NCn * CODESn, 1, Ld);
  k_cbnorm<<<(NCn * CODESn + 255) / 256, 256, 0, stream>>>(cb, cbn);

  k_stage1<<<Bn / 16, 256, 0, stream>>>(x, b1, ln_g, ln_b, b2, W1p, W2p, featp);
  k_router_v<<<Bn / 16, 256, 0, stream>>>(featp, Wkp, bk, chart_q, Wvp, bv, out, v_f);
  k_vq<<<Bn / 16, 256, 0, stream>>>(v_f, cbp, cbn, cb, Ws1p, bs1, Ws2p, bs2, out, vqPart);
  k_finish<<<1, 1, 0, stream>>>(vqPart, out);

  (void)in_sizes; (void)n_in; (void)out_size; (void)ws_size;
}
